// MambaSSM_65824668778588
// MI455X (gfx1250) — matrix operand order repaired and checked by reading, not compiled, not run on hardware
//
#include <hip/hip_runtime.h>
#include <hip/hip_bf16.h>

typedef _Float16 h16;
typedef __attribute__((ext_vector_type(16))) _Float16 v16h;
typedef __attribute__((ext_vector_type(8)))  _Float16 v8h;
typedef __attribute__((ext_vector_type(2)))  _Float16 v2h;
typedef __attribute__((ext_vector_type(8)))  float    v8f;

#define BH   4
#define IMH  96
#define IMW  96
#define HW   (IMH*IMW)          // 9216
#define DMODEL 256
#define DINNER 512
#define DSTATE 64
#define DTRANK 16
#define XDC  160                // x_dbl channel stride (144 padded to 160)
#define LDS_STRIDE 40           // h16 per LDS row: 80B, 16B-aligned, bank-staggered

enum { EP_RELU = 0, EP_LIN = 1, EP_SOFTPLUS = 2, EP_MOD = 3 };

// Wave-private LDS exchange: DS ops of one wave are in-order (ISA 7.1), so no
// block barrier is needed; just stop compiler reordering across the exchange.
__device__ __forceinline__ void wave_sync() {
    __builtin_amdgcn_fence(__ATOMIC_ACQ_REL, "wavefront");
    __builtin_amdgcn_wave_barrier();
    __builtin_amdgcn_fence(__ATOMIC_ACQ_REL, "wavefront");
}

// ---------------------------------------------------------------------------
// Weight prep: f16, B-transposed Wt[cout][k] layouts
// ---------------------------------------------------------------------------
__global__ void k_w_copy(const float* __restrict__ src, h16* __restrict__ dst, int n) {
    int i = blockIdx.x * blockDim.x + threadIdx.x;
    if (i < n) dst[i] = (h16)src[i];
}

__global__ void k_w_copy_pad(const float* __restrict__ src, h16* __restrict__ dst,
                             int nsrc, int ntot) {
    int i = blockIdx.x * blockDim.x + threadIdx.x;
    if (i < ntot) dst[i] = (i < nsrc) ? (h16)src[i] : (h16)0.f;
}

__global__ void k_w_conv(const float* __restrict__ src, h16* __restrict__ dst,
                         int O, int I, int Kh, int Kw) {
    int idx = blockIdx.x * blockDim.x + threadIdx.x;
    int n = O * I * Kh * Kw;
    if (idx >= n) return;
    int K = I * Kh * Kw;
    int o = idx / K;
    int r = idx - o * K;
    int tap = r / I;
    int ci  = r - tap * I;
    int dy = tap / Kw, dx = tap - dy * Kw;
    dst[idx] = (h16)src[((o * I + ci) * Kh + dy) * Kw + dx];
}

__global__ void k_w_deconv(const float* __restrict__ src, h16* __restrict__ dst,
                           int O, int I, int Kh, int Kw) {
    int idx = blockIdx.x * blockDim.x + threadIdx.x;
    int n = O * I * Kh * Kw;
    if (idx >= n) return;
    int K = I * Kh * Kw;
    int o = idx / K;
    int r = idx - o * K;
    int tap = r / I;
    int ci  = r - tap * I;
    int dy = tap / Kw, dx = tap - dy * Kw;
    dst[idx] = (h16)src[((ci * O + o) * Kh + (Kh - 1 - dy)) * Kw + (Kw - 1 - dx)];
}

__global__ void k_w_dtpad(const float* __restrict__ src, h16* __restrict__ dst) {
    int idx = blockIdx.x * blockDim.x + threadIdx.x;
    if (idx >= DINNER * 32) return;
    int o = idx >> 5, k = idx & 31;
    dst[idx] = (k < DTRANK) ? (h16)src[o * DTRANK + k] : (h16)0.f;
}

// ---------------------------------------------------------------------------
// Multi-scale stem: 3->32 convs (k=3/5/7) + relu + concat -> 96ch f16
// ---------------------------------------------------------------------------
__global__ void k_conv_ms(const float* __restrict__ x,
                          const float* __restrict__ w11, const float* __restrict__ b11,
                          const float* __restrict__ w12, const float* __restrict__ b12,
                          const float* __restrict__ w13, const float* __restrict__ b13,
                          h16* __restrict__ out) {
    size_t i = (size_t)blockIdx.x * blockDim.x + threadIdx.x;
    if (i >= (size_t)BH * 96 * HW) return;
    int px = i % IMW;
    int py = (i / IMW) % IMH;
    int c  = (i / HW) % 96;
    int b  = i / ((size_t)HW * 96);
    const float* w; int kk, pad, co; float acc;
    if (c < 32)      { w = w11; kk = 3; pad = 1; co = c;      acc = b11[co]; }
    else if (c < 64) { w = w12; kk = 5; pad = 2; co = c - 32; acc = b12[co]; }
    else             { w = w13; kk = 7; pad = 3; co = c - 64; acc = b13[co]; }
    for (int ci = 0; ci < 3; ++ci)
        for (int dy = 0; dy < kk; ++dy) {
            int yy = py + dy - pad;
            if ((unsigned)yy >= IMH) continue;
            for (int dx = 0; dx < kk; ++dx) {
                int xx = px + dx - pad;
                if ((unsigned)xx >= IMW) continue;
                acc += w[((co * 3 + ci) * kk + dy) * kk + dx] *
                       x[((size_t)(b * 3 + ci) * IMH + yy) * IMW + xx];
            }
        }
    out[i] = (h16)fmaxf(acc, 0.f);
}

// ---------------------------------------------------------------------------
// 1x1 WMMA GEMM: one wave per (16-pixel x 32-outchannel) tile pair.
// A staged once via LDS transpose, reused by 2 WMMAs (2 B fragments).
// ---------------------------------------------------------------------------
__global__ __launch_bounds__(256)
void k_gemm1x1(const h16* __restrict__ act, const h16* __restrict__ wt,
               const float* __restrict__ bias, const h16* __restrict__ act2,
               h16* __restrict__ outp, int Cin, int Cout, int Kred, int mode) {
    __shared__ __align__(16) h16 lds[8][16 * LDS_STRIDE];
    int tid  = threadIdx.x;
    int wv   = tid >> 5;
    int gtid = blockIdx.x * 256 + tid;
    int lane = gtid & 31;
    int half = lane >> 4;
    int mn   = lane & 15;
    int wid  = gtid >> 5;
    int cpairs = Cout >> 5;
    int ct = wid % cpairs;
    int pt = wid / cpairs;
    int p0 = pt * 16;
    int b  = p0 / HW;
    int rem = p0 - b * HW;                 // 16-aligned pixel offset in image
    int n0 = ct * 32 + mn;
    int n1 = n0 + 16;

    const h16* abase = act + (size_t)b * Cin * HW + rem;
    const h16* wrow0 = wt + (size_t)n0 * Kred;
    const h16* wrow1 = wt + (size_t)n1 * Kred;
    h16* lt = lds[wv];
    v8f acc0 = {}, acc1 = {};

    for (int kc = 0; kc < Kred; kc += 32) {
        v16h bfr0 = *(const v16h*)(wrow0 + kc + half * 16);
        v16h bfr1 = *(const v16h*)(wrow1 + kc + half * 16);
        if (kc + 32 < Kred) {
            __builtin_prefetch(wrow0 + kc + 32, 0, 0);
            __builtin_prefetch(wrow1 + kc + 32, 0, 0);
        }
        // stage: lane owns channel kc+lane, loads its 16 pixels (aligned 32B)
        v16h row = *(const v16h*)(abase + (size_t)(kc + lane) * HW);
        wave_sync();
#pragma unroll
        for (int j = 0; j < 16; ++j) lt[j * LDS_STRIDE + lane] = row[j];
        wave_sync();
        const h16* ap = lt + mn * LDS_STRIDE + half * 16;
        union { v16h v; v8h h[2]; } au;
        au.h[0] = *(const v8h*)(ap);
        au.h[1] = *(const v8h*)(ap + 8);
        acc0 = __builtin_amdgcn_wmma_f32_16x16x32_f16(false, au.v, false, bfr0,
                                                      (short)0, acc0, false, false);
        acc1 = __builtin_amdgcn_wmma_f32_16x16x32_f16(false, au.v, false, bfr1,
                                                      (short)0, acc1, false, false);
    }

#pragma unroll
    for (int t = 0; t < 2; ++t) {
        int n = t ? n1 : n0;
        v8f acc = t ? acc1 : acc0;
        float bn = bias ? bias[n] : 0.f;
        size_t obase = ((size_t)(b * Cout + n)) * HW + rem + half * 8;
        v8h ovec;
#pragma unroll
        for (int r = 0; r < 8; ++r) {
            float v = acc[r] + bn;
            if (mode == EP_RELU) {
                v = fmaxf(v, 0.f);
            } else if (mode == EP_SOFTPLUS) {
                v = (v > 20.f) ? v : log1pf(__expf(v));
            } else if (mode == EP_MOD) {
                float g = 1.f / (1.f + __expf(-v));
                v = (float)act2[obase + r] * g;
            }
            ovec[r] = (h16)v;
        }
        *(v8h*)(outp + obase) = ovec;
    }
}

// ---------------------------------------------------------------------------
// 3x3/pad1 WMMA implicit-GEMM conv (+optional skip add), relu epilogue.
// Aligned center row + one scalar halo + register shift per tap; same LDS
// staging, 2 cout tiles per wave.
// ---------------------------------------------------------------------------
__global__ __launch_bounds__(256)
void k_gemm3x3(const h16* __restrict__ act, const h16* __restrict__ skip,
               const h16* __restrict__ wt, const float* __restrict__ bias,
               h16* __restrict__ outp, int Cin, int Cout) {
    __shared__ __align__(16) h16 lds[8][16 * LDS_STRIDE];
    int tid  = threadIdx.x;
    int wv   = tid >> 5;
    int gtid = blockIdx.x * 256 + tid;
    int lane = gtid & 31;
    int half = lane >> 4;
    int mn   = lane & 15;
    int wid  = gtid >> 5;
    int cpairs = Cout >> 5;
    int ct = wid % cpairs;
    int pt = wid / cpairs;
    int p0 = pt * 16;
    int b  = p0 / HW;
    int rem = p0 - b * HW;
    int py = rem / IMW;
    int px = rem - py * IMW;               // multiple of 16
    int n0 = ct * 32 + mn;
    int n1 = n0 + 16;

    const int K = 9 * Cin;
    const h16* wrow0 = wt + (size_t)n0 * K;
    const h16* wrow1 = wt + (size_t)n1 * K;
    h16* lt = lds[wv];
    v8f acc0 = {}, acc1 = {};

#pragma unroll
    for (int tap = 0; tap < 9; ++tap) {
        const int dy = tap / 3 - 1, dx = tap % 3 - 1;
        int yy = py + dy;
        bool rowok = (unsigned)yy < IMH;           // uniform over wave
        const h16* rbase = act + ((size_t)b * Cin) * HW + yy * IMW + px;
        const h16* sbase = skip ? skip + ((size_t)b * Cin) * HW + yy * IMW + px : nullptr;

        for (int kc = 0; kc < Cin; kc += 32) {
            int kg = tap * Cin + kc;
            v16h bfr0 = *(const v16h*)(wrow0 + kg + half * 16);
            v16h bfr1 = *(const v16h*)(wrow1 + kg + half * 16);
            v16h row = {};
            if (rowok) {
                size_t coff = (size_t)(kc + lane) * HW;
                v16h ctr = *(const v16h*)(rbase + coff);
                if (skip) {
                    v16h sk = *(const v16h*)(sbase + coff);
#pragma unroll
                    for (int j = 0; j < 16; ++j) ctr[j] = ctr[j] + sk[j];
                }
                if (dx == 0) {
                    row = ctr;
                } else if (dx < 0) {
                    h16 lft = (h16)0.f;
                    if (px > 0) {
                        lft = rbase[coff - 1];
                        if (skip) lft = lft + sbase[coff - 1];
                    }
#pragma unroll
                    for (int j = 15; j >= 1; --j) row[j] = ctr[j - 1];
                    row[0] = lft;
                } else {
                    h16 rgt = (h16)0.f;
                    if (px + 16 < IMW) {
                        rgt = rbase[coff + 16];
                        if (skip) rgt = rgt + sbase[coff + 16];
                    }
#pragma unroll
                    for (int j = 0; j < 15; ++j) row[j] = ctr[j + 1];
                    row[15] = rgt;
                }
            }
            wave_sync();
#pragma unroll
            for (int j = 0; j < 16; ++j) lt[j * LDS_STRIDE + lane] = row[j];
            wave_sync();
            const h16* ap = lt + mn * LDS_STRIDE + half * 16;
            union { v16h v; v8h h[2]; } au;
            au.h[0] = *(const v8h*)(ap);
            au.h[1] = *(const v8h*)(ap + 8);
            acc0 = __builtin_amdgcn_wmma_f32_16x16x32_f16(false, au.v, false, bfr0,
                                                          (short)0, acc0, false, false);
            acc1 = __builtin_amdgcn_wmma_f32_16x16x32_f16(false, au.v, false, bfr1,
                                                          (short)0, acc1, false, false);
        }
    }

#pragma unroll
    for (int t = 0; t < 2; ++t) {
        int n = t ? n1 : n0;
        v8f acc = t ? acc1 : acc0;
        float bn = bias[n];
        size_t obase = ((size_t)(b * Cout + n)) * HW + py * IMW + px + half * 8;
        v8h ovec;
#pragma unroll
        for (int r = 0; r < 8; ++r) ovec[r] = (h16)fmaxf(acc[r] + bn, 0.f);
        *(v8h*)(outp + obase) = ovec;
    }
}

// ---------------------------------------------------------------------------
// Depthwise causal conv1d (d_conv=4) + bias + SiLU on xz[:, :512, :]
// ---------------------------------------------------------------------------
__global__ void k_dwconv(const h16* __restrict__ xz, const float* __restrict__ cw,
                         const float* __restrict__ cb, h16* __restrict__ u) {
    size_t i = (size_t)blockIdx.x * blockDim.x + threadIdx.x;
    if (i >= (size_t)BH * DINNER * HW) return;
    int l = i % HW;
    int c = (i / HW) % DINNER;
    int b = i / ((size_t)HW * DINNER);
    const h16* src = xz + ((size_t)(b * 2 * DINNER + c)) * HW;
    float acc = cb[c];
#pragma unroll
    for (int t = 0; t < 4; ++t) {
        int ll = l - 3 + t;
        if (ll >= 0) acc += cw[c * 4 + t] * (float)src[ll];
    }
    u[i] = (h16)(acc / (1.f + __expf(-acc)));
}

// ---------------------------------------------------------------------------
// Selective scan: one wave32 per (b, d); lane owns states s=lane and s=lane+32.
// Unrolled x2 with paired 32-bit loads + single 32-bit shfl broadcasts.
// ---------------------------------------------------------------------------
__global__ __launch_bounds__(256)
void k_scan(const h16* __restrict__ dt, const h16* __restrict__ u,
            const h16* __restrict__ xdbl, const h16* __restrict__ xz,
            const float* __restrict__ A_log, const float* __restrict__ Dp,
            h16* __restrict__ ymod) {
    int gw   = (blockIdx.x * blockDim.x + threadIdx.x) >> 5;   // [0, BH*DINNER)
    int lane = threadIdx.x & 31;
    int b = gw / DINNER;
    int d = gw - b * DINNER;

    float a0 = -__expf(A_log[d * DSTATE + lane]);
    float a1 = -__expf(A_log[d * DSTATE + 32 + lane]);
    float h0 = 0.f, h1 = 0.f;

    const h16* dtp = dt + ((size_t)(b * DINNER + d)) * HW;
    const h16* up  = u  + ((size_t)(b * DINNER + d)) * HW;
    const h16* zp  = xz + ((size_t)(b * 2 * DINNER + DINNER + d)) * HW;
    const h16* Bp0 = xdbl + ((size_t)(b * XDC + DTRANK + lane)) * HW;
    const h16* Bp1 = Bp0 + (size_t)32 * HW;
    const h16* Cp0 = xdbl + ((size_t)(b * XDC + DTRANK + DSTATE + lane)) * HW;
    const h16* Cp1 = Cp0 + (size_t)32 * HW;
    h16* yp = ymod + ((size_t)(b * DINNER + d)) * HW;
    float Dd = Dp[d];

    union pun { int i; v2h h; };
    for (int l = 0; l < HW; l += 2) {
        pun dt2, uv2, zv2;
        dt2.i = uv2.i = zv2.i = 0;
        if (lane == 0) {
            dt2.i = *(const int*)(dtp + l);
            uv2.i = *(const int*)(up + l);
            zv2.i = *(const int*)(zp + l);
        }
        dt2.i = __shfl(dt2.i, 0, 32);
        uv2.i = __shfl(uv2.i, 0, 32);
        pun B0, B1, C0, C1;
        B0.h = *(const v2h*)(Bp0 + l);
        B1.h = *(const v2h*)(Bp1 + l);
        C0.h = *(const v2h*)(Cp0 + l);
        C1.h = *(const v2h*)(Cp1 + l);

        pun yv;
#pragma unroll
        for (int q = 0; q < 2; ++q) {
            float dtv = (float)dt2.h[q];
            float uv  = (float)uv2.h[q];
            h0 = __expf(dtv * a0) * h0 + dtv * (float)B0.h[q] * uv;
            h1 = __expf(dtv * a1) * h1 + dtv * (float)B1.h[q] * uv;
            float y = h0 * (float)C0.h[q] + h1 * (float)C1.h[q];
#pragma unroll
            for (int off = 16; off; off >>= 1) y += __shfl_xor(y, off, 32);
            if (lane == 0) {
                float zv = (float)zv2.h[q];
                float sil = zv / (1.f + __expf(-zv));
                yv.h[q] = (h16)((y + uv * Dd) * sil);
            }
        }
        if (lane == 0) *(int*)(yp + l) = yv.i;
    }
}

// ---------------------------------------------------------------------------
// Final deconv 64->3 (+skip) with sigmoid, f32 output
// ---------------------------------------------------------------------------
__global__ void k_conv_final(const h16* __restrict__ x5, const h16* __restrict__ x1r,
                             const float* __restrict__ d3w, const float* __restrict__ d3b,
                             float* __restrict__ out) {
    size_t i = (size_t)blockIdx.x * blockDim.x + threadIdx.x;
    if (i >= (size_t)BH * 3 * HW) return;
    int px = i % IMW;
    int py = (i / IMW) % IMH;
    int o  = (i / HW) % 3;
    int b  = i / ((size_t)HW * 3);
    float acc = d3b[o];
    for (int ci = 0; ci < 64; ++ci)
        for (int dy = 0; dy < 3; ++dy) {
            int yy = py + dy - 1;
            if ((unsigned)yy >= IMH) continue;
            for (int dx = 0; dx < 3; ++dx) {
                int xx = px + dx - 1;
                if ((unsigned)xx >= IMW) continue;
                float wk = d3w[((ci * 3 + o) * 3 + (2 - dy)) * 3 + (2 - dx)];
                size_t idx = ((size_t)(b * 64 + ci)) * HW + yy * IMW + xx;
                acc += wk * ((float)x5[idx] + (float)x1r[idx]);
            }
        }
    out[i] = 1.f / (1.f + __expf(-acc));
}

// ---------------------------------------------------------------------------
extern "C" void kernel_launch(void* const* d_in, const int* in_sizes, int n_in,
                              void* d_out, int out_size, void* d_ws, size_t ws_size,
                              hipStream_t stream) {
    const float* x    = (const float*)d_in[0];
    const float* w11  = (const float*)d_in[1];   const float* b11 = (const float*)d_in[2];
    const float* w12  = (const float*)d_in[3];   const float* b12 = (const float*)d_in[4];
    const float* w13  = (const float*)d_in[5];   const float* b13 = (const float*)d_in[6];
    const float* wred = (const float*)d_in[7];   const float* bred = (const float*)d_in[8];
    const float* w2   = (const float*)d_in[9];   const float* b2  = (const float*)d_in[10];
    const float* w3   = (const float*)d_in[11];  const float* b3  = (const float*)d_in[12];
    const float* in_w = (const float*)d_in[13];
    const float* cw   = (const float*)d_in[14];  const float* cb  = (const float*)d_in[15];
    const float* xp_w = (const float*)d_in[16];
    const float* dtw  = (const float*)d_in[17];  const float* dtb = (const float*)d_in[18];
    const float* A_log= (const float*)d_in[19];  const float* Dpv = (const float*)d_in[20];
    const float* out_w= (const float*)d_in[21];
    const float* fm_w = (const float*)d_in[22];  const float* fm_b = (const float*)d_in[23];
    const float* d1w  = (const float*)d_in[24];  const float* d1b = (const float*)d_in[25];
    const float* d2w  = (const float*)d_in[26];  const float* d2b = (const float*)d_in[27];
    const float* d3w  = (const float*)d_in[28];  const float* d3b = (const float*)d_in[29];

    h16* ws = (h16*)d_ws;
    size_t off = 0;
    auto alloc = [&](size_t nelem) { h16* p = ws + off; off += nelem; return p; };

    h16* wt_red = alloc(64  * 96);
    h16* wt_2   = alloc(128 * 576);
    h16* wt_3   = alloc(256 * 1152);
    h16* wt_in  = alloc(1024 * 256);
    h16* wt_xp  = alloc(XDC * 512);          // padded 144 -> 160 rows
    h16* wt_dt  = alloc(512 * 32);
    h16* wt_out = alloc(256 * 512);
    h16* wt_fm  = alloc(256 * 256);
    h16* wt_d1  = alloc(128 * (9 * 256));
    h16* wt_d2  = alloc(64  * (9 * 128));
    h16* xcat = alloc((size_t)BH * 96   * HW);
    h16* x1r  = alloc((size_t)BH * 64   * HW);
    h16* xb2  = alloc((size_t)BH * 128  * HW);
    h16* xb3  = alloc((size_t)BH * 256  * HW);
    h16* xz   = alloc((size_t)BH * 1024 * HW);
    h16* uu   = alloc((size_t)BH * 512  * HW);
    h16* xdbl = alloc((size_t)BH * XDC  * HW);
    h16* dtb_ = alloc((size_t)BH * 512  * HW);
    h16* ymod = alloc((size_t)BH * 512  * HW);
    h16* x3s  = alloc((size_t)BH * 256  * HW);
    h16* x3sm = alloc((size_t)BH * 256  * HW);
    h16* x4   = alloc((size_t)BH * 128  * HW);
    h16* x5   = alloc((size_t)BH * 64   * HW);

    auto cdiv = [](size_t a, size_t b) { return (int)((a + b - 1) / b); };

    // weight prep
    k_w_copy<<<cdiv(64*96,256),256,0,stream>>>(wred, wt_red, 64*96);
    k_w_conv<<<cdiv(128*576,256),256,0,stream>>>(w2, wt_2, 128, 64, 3, 3);
    k_w_conv<<<cdiv(256*1152,256),256,0,stream>>>(w3, wt_3, 256, 128, 3, 3);
    k_w_copy<<<cdiv(1024*256,256),256,0,stream>>>(in_w, wt_in, 1024*256);
    k_w_copy_pad<<<cdiv(XDC*512,256),256,0,stream>>>(xp_w, wt_xp, 144*512, XDC*512);
    k_w_dtpad<<<cdiv(512*32,256),256,0,stream>>>(dtw, wt_dt);
    k_w_copy<<<cdiv(256*512,256),256,0,stream>>>(out_w, wt_out, 256*512);
    k_w_copy<<<cdiv(256*256,256),256,0,stream>>>(fm_w, wt_fm, 256*256);
    k_w_deconv<<<cdiv(128*9*256,256),256,0,stream>>>(d1w, wt_d1, 128, 256, 3, 3);
    k_w_deconv<<<cdiv(64*9*128,256),256,0,stream>>>(d2w, wt_d2, 64, 128, 3, 3);

    // encoder stem
    k_conv_ms<<<cdiv((size_t)BH*96*HW,256),256,0,stream>>>(x, w11,b11, w12,b12, w13,b13, xcat);

    auto g11 = [&](const h16* act, const h16* wt, const float* bias, const h16* act2,
                   h16* outp, int Cin, int Cout, int Kred, int mode) {
        int tiles = (BH * HW / 16) * (Cout / 32);
        k_gemm1x1<<<tiles / 8, 256, 0, stream>>>(act, wt, bias, act2, outp,
                                                 Cin, Cout, Kred, mode);
    };
    auto g33 = [&](const h16* act, const h16* skip, const h16* wt, const float* bias,
                   h16* outp, int Cin, int Cout) {
        int tiles = (BH * HW / 16) * (Cout / 32);
        k_gemm3x3<<<tiles / 8, 256, 0, stream>>>(act, skip, wt, bias, outp, Cin, Cout);
    };

    g11(xcat, wt_red, bred, nullptr, x1r, 96, 64, 96, EP_RELU);
    g33(x1r,  nullptr, wt_2, b2, xb2, 64, 128);
    g33(xb2,  nullptr, wt_3, b3, xb3, 128, 256);

    // mamba
    g11(xb3,  wt_in,  nullptr, nullptr, xz,   256, 1024, 256, EP_LIN);
    k_dwconv<<<cdiv((size_t)BH*512*HW,256),256,0,stream>>>(xz, cw, cb, uu);
    g11(uu,   wt_xp,  nullptr, nullptr, xdbl, 512, XDC, 512, EP_LIN);
    g11(xdbl, wt_dt,  dtb,     nullptr, dtb_, XDC, 512, 32, EP_SOFTPLUS);
    k_scan<<<(BH * DINNER * 32) / 256, 256, 0, stream>>>(dtb_, uu, xdbl, xz, A_log, Dpv, ymod);
    g11(ymod, wt_out, nullptr, nullptr, x3s,  512, 256, 512, EP_LIN);

    // modulation + decoder
    g11(x3s,  wt_fm, fm_b, x3s, x3sm, 256, 256, 256, EP_MOD);
    g33(x3sm, nullptr, wt_d1, d1b, x4, 256, 128);
    g33(x4,   xb2,     wt_d2, d2b, x5, 128, 64);

    k_conv_final<<<cdiv((size_t)BH*3*HW,256),256,0,stream>>>(x5, x1r, d3w, d3b, (float*)d_out);
}
